// Attention_82197084111640
// MI455X (gfx1250) — matrix operand order repaired and checked by reading, not compiled, not run on hardware
//
#include <hip/hip_runtime.h>
#include <stdint.h>

#define H    32
#define KVH  8
#define G    4
#define D    128
#define DV   128
#define SEQ  1024
#define SCALE 0.08838834764831845f
#define NEGF (-1e30f)

typedef __attribute__((ext_vector_type(16))) _Float16 v16h;
typedef __attribute__((ext_vector_type(8)))  float    v8f;

// LDS layout: [buf 0..1][kv 0..1][row 0..31][128 f32], XOR-swizzled in 4-float
// groups: logical group g of row r lives at physical group (g ^ r) & 31.
// Row stride 512B (16B aligned); swizzle kills the 64-bank row-stride aliasing.
__device__ __forceinline__ int sw_idx(int buf, int kv, int row, int g) {
    return buf * 8192 + kv * 4096 + row * 128 + ((g ^ (row & 31)) & 31) * 4;
}

__device__ __forceinline__ void async_cp16B(uint32_t lds_addr, const float* gptr) {
    // GLOBAL_LOAD_ASYNC_TO_LDS_B128: VDST = LDS byte address, VADDR = 64b global
    asm volatile("global_load_async_to_lds_b128 %0, %1, off"
                 :: "v"(lds_addr), "v"(gptr) : "memory");
}

__device__ __forceinline__ v16h frag16(float4 a0, float4 a1, float4 a2, float4 a3) {
    v16h f;
    f[0]  = (_Float16)a0.x; f[1]  = (_Float16)a0.y; f[2]  = (_Float16)a0.z; f[3]  = (_Float16)a0.w;
    f[4]  = (_Float16)a1.x; f[5]  = (_Float16)a1.y; f[6]  = (_Float16)a1.z; f[7]  = (_Float16)a1.w;
    f[8]  = (_Float16)a2.x; f[9]  = (_Float16)a2.y; f[10] = (_Float16)a2.z; f[11] = (_Float16)a2.w;
    f[12] = (_Float16)a3.x; f[13] = (_Float16)a3.y; f[14] = (_Float16)a3.z; f[15] = (_Float16)a3.w;
    return f;
}

__global__ __launch_bounds__(256)
void fa_fwd_gqa_kernel(const float* __restrict__ Q,
                       const float* __restrict__ Kp,
                       const float* __restrict__ Vp,
                       float* __restrict__ O)
{
    __shared__ float sh[2 * 2 * 32 * 128];   // 64 KB, double-buffered K/V (f32)

    const int lane = threadIdx.x & 31;
    const int wave = threadIdx.x >> 5;
    const int qb   = blockIdx.x;
    const int kvh  = blockIdx.y;
    const int b    = blockIdx.z;

    const int head  = kvh * G + (wave & 3);
    const int qtile = qb * 32 + (wave >> 2) * 16;
    const int lg    = lane >> 4;
    const int ln    = lane & 15;

    // ---- Q^T B-fragments (lane = q, halves = d), scale folded, in regs ----
    v16h qf[4];
    {
        const float* qbase = Q + (size_t)(b * SEQ + qtile) * (H * D) + head * D;
        #pragma unroll
        for (int c = 0; c < 4; ++c) {
            const float* p = qbase + (size_t)ln * (H * D) + c * 32 + 8 * lg;
            #pragma unroll
            for (int h = 0; h < 16; ++h)
                qf[c][h] = (_Float16)(p[(h < 8) ? h : (h + 8)] * SCALE);
        }
    }

    v8f of[8];
    v8f vzero = {};
    #pragma unroll
    for (int j = 0; j < 8; ++j) of[j] = vzero;
    float m_run = NEGF, l_run = 0.0f;

    // ---- async K/V stager: 256 threads x (4 K + 4 V) b128 copies per block ----
    const int srow = threadIdx.x >> 3;          // 0..31 (key row)
    const int sgb  = (threadIdx.x & 7) * 4;     // first of 4 logical groups
    const size_t kstride = (size_t)(KVH * D);
    #define ISSUE_BLOCK(kb_, buf_)                                                \
        do {                                                                      \
            const size_t tok_ = (size_t)(b * SEQ + (kb_) + srow);                 \
            const float* kg_ = Kp + tok_ * kstride + kvh * D;                     \
            const float* vg_ = Vp + tok_ * kstride + kvh * DV;                    \
            _Pragma("unroll")                                                     \
            for (int i_ = 0; i_ < 4; ++i_) {                                      \
                const int g_ = sgb + i_;                                          \
                async_cp16B((uint32_t)(uintptr_t)&sh[sw_idx((buf_), 0, srow, g_)],\
                            kg_ + g_ * 4);                                        \
                async_cp16B((uint32_t)(uintptr_t)&sh[sw_idx((buf_), 1, srow, g_)],\
                            vg_ + g_ * 4);                                        \
            }                                                                     \
        } while (0)

    const int nblk = qb + 1;
    ISSUE_BLOCK(0, 0);                          // prologue: prefetch block 0

    for (int blk = 0; blk < nblk; ++blk) {
        const int cur = blk & 1;
        const int kb  = blk * 32;

        if (blk + 1 < nblk) {
            ISSUE_BLOCK(kb + 32, cur ^ 1);      // prefetch next block
            asm volatile("s_wait_asynccnt 8" ::: "memory");   // oldest 8 (cur) done
        } else {
            asm volatile("s_wait_asynccnt 0" ::: "memory");
        }
        __syncthreads();                        // cur buffer visible to all waves

        // ---- S^T = K * (Q*scale)^T : A = K-tile (lane = key row) ----
        float s[2][8];
        #pragma unroll
        for (int t = 0; t < 2; ++t) {
            const int r = t * 16 + ln;
            v8f acc = vzero;
            #pragma unroll
            for (int c = 0; c < 4; ++c) {
                const int g0 = c * 8 + 2 * lg;        // halves 0-7:  d = c*32+8lg+0..7
                const int g1 = c * 8 + 4 + 2 * lg;    // halves 8-15: d = c*32+16+8lg+0..7
                float4 a0 = *(const float4*)&sh[sw_idx(cur, 0, r, g0)];
                float4 a1 = *(const float4*)&sh[sw_idx(cur, 0, r, g0 + 1)];
                float4 a2 = *(const float4*)&sh[sw_idx(cur, 0, r, g1)];
                float4 a3 = *(const float4*)&sh[sw_idx(cur, 0, r, g1 + 1)];
                v16h kf = frag16(a0, a1, a2, a3);
                acc = __builtin_amdgcn_wmma_f32_16x16x32_f16(
                        false, kf, false, qf[c], (short)0, acc, false, false);
            }
            const int q_g = qtile + ln;
            #pragma unroll
            for (int v = 0; v < 8; ++v) {
                const int key_g = kb + t * 16 + v + 8 * lg;
                s[t][v] = (key_g <= q_g) ? acc[v] : NEGF;
            }
        }

        // ---- online softmax (per-lane + partner-lane combine) ----
        float mloc = NEGF;
        #pragma unroll
        for (int t = 0; t < 2; ++t)
            #pragma unroll
            for (int v = 0; v < 8; ++v) mloc = fmaxf(mloc, s[t][v]);
        mloc = fmaxf(mloc, __shfl_xor(mloc, 16, 32));
        const float mnew = fmaxf(m_run, mloc);
        const float corr = __expf(m_run - mnew);

        v16h pf;                                  // P lands in A-operand layout
        float rsum = 0.0f;
        #pragma unroll
        for (int t = 0; t < 2; ++t)
            #pragma unroll
            for (int v = 0; v < 8; ++v) {
                const float p = __expf(s[t][v] - mnew);
                rsum += p;
                pf[t * 8 + v] = (_Float16)p;
            }
        rsum += __shfl_xor(rsum, 16, 32);
        l_run = l_run * corr + rsum;
        m_run = mnew;

        // ---- rescale O accumulators ----
        float fv[8];
        #pragma unroll
        for (int v = 0; v < 8; ++v)
            fv[v] = __shfl(corr, v + 8 * lg, 32);
        #pragma unroll
        for (int j = 0; j < 8; ++j)
            #pragma unroll
            for (int v = 0; v < 8; ++v) of[j][v] *= fv[v];

        // ---- O += P * V : B = V chunk (lane = dv, halves = key row) ----
        #pragma unroll
        for (int j = 0; j < 8; ++j) {
            v16h vf;
            #pragma unroll
            for (int e = 0; e < 16; ++e) {
                const int key = 8 * lg + ((e < 8) ? e : (e + 8));
                vf[e] = (_Float16)sh[sw_idx(cur, 1, key, (j * 16 + ln) >> 2) + (ln & 3)];
            }
            of[j] = __builtin_amdgcn_wmma_f32_16x16x32_f16(
                      false, pf, false, vf, (short)0, of[j], false, false);
        }
        __syncthreads();   // all reads of cur done before it is refilled next iter
    }

    // ---- finalize: divide by l, store ----
    const size_t obase = (size_t)(b * SEQ + qtile) * (H * DV) + head * DV;
    #pragma unroll
    for (int v = 0; v < 8; ++v) {
        const float lv  = __shfl(l_run, v + 8 * lg, 32);
        const float inv = 1.0f / lv;
        float* orow = O + obase + (size_t)(v + 8 * lg) * (H * DV);
        #pragma unroll
        for (int j = 0; j < 8; ++j)
            orow[j * 16 + ln] = of[j][v] * inv;
    }
}

extern "C" void kernel_launch(void* const* d_in, const int* in_sizes, int n_in,
                              void* d_out, int out_size, void* d_ws, size_t ws_size,
                              hipStream_t stream) {
    const float* Q = (const float*)d_in[0];
    const float* K = (const float*)d_in[1];
    const float* V = (const float*)d_in[2];
    float* O = (float*)d_out;

    const int T = in_sizes[0] / (H * D);
    const int B = T / SEQ;
    dim3 grid(SEQ / 32, KVH, B);
    fa_fwd_gqa_kernel<<<grid, 256, 0, stream>>>(Q, K, V, O);
}
